// RiemannFormerAttention_59450937311385
// MI455X (gfx1250) — matrix operand order repaired and checked by reading, not compiled, not run on hardware
//
#include <hip/hip_runtime.h>
#include <hip/hip_bf16.h>

typedef __attribute__((ext_vector_type(16))) _Float16 v16h;
typedef __attribute__((ext_vector_type(8)))  _Float16 v8h;
typedef __attribute__((ext_vector_type(8)))  float    v8f;

#define WMMA_F16(a, b, c) \
    __builtin_amdgcn_wmma_f32_16x16x32_f16(false, (a), false, (b), (short)0, (c), false, false)

__device__ __forceinline__ v16h cat8(v8h lo, v8h hi) {
    return __builtin_shufflevector(lo, hi, 0,1,2,3,4,5,6,7,8,9,10,11,12,13,14,15);
}

// A-fragment (16x32 f16). src points at (row0, k0), row-major, leading dim lda.
__device__ __forceinline__ v16h load_a_frag(const _Float16* src, int lda, int lane) {
    const int row  = lane & 15;
    const int half = lane >> 4;
    const _Float16* p = src + (size_t)row * lda;
    v8h lo = *(const v8h*)(p + half * 8);
    v8h hi = *(const v8h*)(p + 16 + half * 8);
    return cat8(lo, hi);
}

// B-fragment (32x16 f16), column n contiguous with ldc elements between columns.
__device__ __forceinline__ v16h load_b_frag(const _Float16* colbase, int ldc, int lane) {
    const _Float16* p = colbase + (size_t)(lane & 15) * ldc + (lane >> 4) * 8;
    v8h lo = *(const v8h*)p;
    v8h hi = *(const v8h*)(p + 16);
    return cat8(lo, hi);
}

// ---- CDNA5 async global->LDS copy (ASYNCcnt path), one b128 per active lane.
// LDS byte address = low 32 bits of the generic pointer (ISA 10.2 aperture rule).
__device__ __forceinline__ void async_load_b128(void* lds, const void* gptr) {
    uint32_t l = (uint32_t)(uintptr_t)lds;
    uint64_t g = (uint64_t)(uintptr_t)gptr;
    asm volatile("global_load_async_to_lds_b128 %0, %1, off"
                 :: "v"(l), "v"(g) : "memory");
}
__device__ __forceinline__ void wait_async0() {
    asm volatile("s_wait_asynccnt 0x0" ::: "memory");
}
__device__ __forceinline__ void wait_async1() {
    asm volatile("s_wait_asynccnt 0x1" ::: "memory");
}
__device__ __forceinline__ void wait_async2() {
    asm volatile("s_wait_asynccnt 0x2" ::: "memory");
}

// ---------------------------------------------------------------- conversions
__global__ void cvt_f32_to_f16(const float* __restrict__ in, _Float16* __restrict__ out, int n) {
    int i = blockIdx.x * blockDim.x + threadIdx.x;
    if (i < n) out[i] = (_Float16)in[i];
}

// Wt[n][k] = (f16) W[k][n]; block (32,8), grid (D/32, D/32).
__global__ __launch_bounds__(256) void transpose_w_f16(const float* __restrict__ W,
                                                       _Float16* __restrict__ Wt, int Dm) {
    __shared__ float tile[32][33];
    const int k0 = blockIdx.x * 32, n0 = blockIdx.y * 32;
    const int tx = threadIdx.x, ty = threadIdx.y;
    for (int r = ty; r < 32; r += 8)
        tile[r][tx] = W[(size_t)(k0 + r) * Dm + n0 + tx];
    __syncthreads();
    for (int r = ty; r < 32; r += 8)
        Wt[(size_t)(n0 + r) * Dm + k0 + tx] = (_Float16)tile[tx][r];
}

// Vt[((b*H+h)*64+dv)*L + l] = (f16) Vf[(b*L+l)*D + h*64 + dv]
__global__ void vt_transpose(const float* __restrict__ Vf, _Float16* __restrict__ Vt,
                             int Bx, int Lx, int Hh) {
    const int Dm = Hh * 64;
    int idx = blockIdx.x * blockDim.x + threadIdx.x;
    if (idx >= Bx * Lx * Dm) return;
    int dfull = idx % Dm;
    int l = (idx / Dm) % Lx;
    int b = idx / (Dm * Lx);
    int h = dfull >> 6, dv = dfull & 63;
    Vt[((size_t)(b * Hh + h) * 64 + dv) * Lx + l] = (_Float16)Vf[idx];
}

// ------------------------------------------------------------------ WMMA GEMM
// C[M][N] = A[M][K](f16) * Bt[N][K](f16) + bias[N].  Block = 8 waves stacked
// along M sharing one async-staged, double-buffered 64x32 B tile in LDS.
// Requires M % 128 == 0, N % 64 == 0, K % 32 == 0.
__global__ __launch_bounds__(256) void gemm_f16_wmma(const _Float16* __restrict__ A,
                                                     const _Float16* __restrict__ Bt,
                                                     const float* __restrict__ bias,
                                                     float* __restrict__ C,
                                                     int M, int N, int K) {
    __shared__ _Float16 smB[2][64 * 32];  // [buf][n][k]

    const int tid  = threadIdx.x;
    const int lane = tid & 31;
    const int wib  = tid >> 5;
    const int tilesN = N >> 6;
    const int n0 = (blockIdx.x % tilesN) * 64;
    const int m0 = ((blockIdx.x / tilesN) * 8 + wib) * 16;
    const int half = lane >> 4;

    auto fillB = [&](int p, int k0) {
        int row = tid >> 2, chunk = (tid & 3) * 8;      // 64 rows x 4 b128 chunks
        async_load_b128(&smB[p][row * 32 + chunk],
                        Bt + (size_t)(n0 + row) * K + k0 + chunk);
    };

    fillB(0, 0);
    v8f acc[4] = {{}, {}, {}, {}};
    int par = 0;
    for (int k0 = 0; k0 < K; k0 += 32) {
        const bool pre = (k0 + 32) < K;
        if (pre) {
            fillB(par ^ 1, k0 + 32);
            __builtin_prefetch(A + (size_t)m0 * K + k0 + 64, 0, 3);
            wait_async1();                 // older fill (current buf) complete
        } else {
            wait_async0();
        }
        __syncthreads();
        v16h a = load_a_frag(A + (size_t)m0 * K + k0, K, lane);
#pragma unroll
        for (int c = 0; c < 4; ++c) {
            v16h b = load_b_frag(&smB[par][(c * 16) * 32], 32, lane);
            acc[c] = WMMA_F16(a, b, acc[c]);
        }
        __syncthreads();                   // buf consumed before next overwrite
        par ^= 1;
    }
#pragma unroll
    for (int c = 0; c < 4; ++c)
#pragma unroll
        for (int v = 0; v < 8; ++v) {
            int m = m0 + v + 8 * half;
            int n = n0 + c * 16 + (lane & 15);
            C[(size_t)m * N + n] = acc[c][v] + bias[n];
        }
}

// ------------------------------------------------------- Cayley transform
// One 64-thread block per (h,l): Gauss-Jordan solve of
// (I - 0.5*pos*X) T0 = (I + 0.5*pos*X)  (valid w/o pivoting for I - skew),
// T = exp(-0.5*log_s)*T0, then Qr/Kr[b,h,l,i] = sum_j T[j][i]*Q/K[b,l,h*64+j].
__global__ __launch_bounds__(64) void cayley_apply(const float* __restrict__ Xg,
                                                   const int* __restrict__ positions,
                                                   const float* __restrict__ log_s,
                                                   const float* __restrict__ Qf,
                                                   const float* __restrict__ Kf,
                                                   _Float16* __restrict__ Qr,
                                                   _Float16* __restrict__ Kr,
                                                   int Bx, int Lx, int Hh) {
    __shared__ float Ms[64 * 65];
    __shared__ float Ns[64 * 65];
    __shared__ float col[64];
    __shared__ float vec[64];

    const int t  = threadIdx.x;
    const int h  = blockIdx.x / Lx;
    const int l  = blockIdx.x % Lx;
    const int Dm = Hh * 64;
    const float s = 0.5f * (float)positions[l];

    const float* Xh = Xg + (size_t)h * 64 * 64;
    for (int r = 0; r < 64; ++r) {
        float xv  = Xh[r * 64 + t];
        float eye = (r == t) ? 1.0f : 0.0f;
        Ms[r * 65 + t] = eye - s * xv;
        Ns[r * 65 + t] = eye + s * xv;
    }
    __syncthreads();

    for (int k = 0; k < 64; ++k) {
        col[t] = Ms[t * 65 + k];
        __syncthreads();
        float inv = 1.0f / col[k];
        float mk = Ms[k * 65 + t] * inv;
        float nk = Ns[k * 65 + t] * inv;
        Ms[k * 65 + t] = mk;
        Ns[k * 65 + t] = nk;
        __syncthreads();
        for (int i = 0; i < 64; ++i) {
            if (i == k) continue;
            float f = col[i];
            Ms[i * 65 + t] -= f * mk;
            Ns[i * 65 + t] -= f * nk;
        }
        __syncthreads();
    }

    const float scale = __expf(-0.5f * log_s[h]);
    for (int b = 0; b < Bx; ++b) {
        vec[t] = Qf[(size_t)(b * Lx + l) * Dm + h * 64 + t];
        __syncthreads();
        float accq = 0.f;
        for (int j = 0; j < 64; ++j) accq += Ns[j * 65 + t] * vec[j];
        Qr[((size_t)(b * Hh + h) * Lx + l) * 64 + t] = (_Float16)(scale * accq);
        __syncthreads();

        vec[t] = Kf[(size_t)(b * Lx + l) * Dm + h * 64 + t];
        __syncthreads();
        float acck = 0.f;
        for (int j = 0; j < 64; ++j) acck += Ns[j * 65 + t] * vec[j];
        Kr[((size_t)(b * Hh + h) * Lx + l) * 64 + t] = (_Float16)(scale * acck);
        __syncthreads();
    }
}

// --------------------------------------------------------- flash attention
// Block = 8 waves = 128 consecutive query rows of one (b,h).  32-key K/V
// tiles are async-staged to LDS (double-buffered) and shared by all waves.
__global__ __launch_bounds__(256) void attn_wmma(const _Float16* __restrict__ Qr,
                                                 const _Float16* __restrict__ Kr,
                                                 const _Float16* __restrict__ Vt,
                                                 const float* __restrict__ log_sigma,
                                                 _Float16* __restrict__ O,
                                                 int Bx, int Lx, int Hh) {
    __shared__ _Float16 smK[2][32 * 64];  // [buf][key j][feat]
    __shared__ _Float16 smV[2][64 * 32];  // [buf][dv][key j]
    __shared__ _Float16 smP[8][16 * 32];  // per-wave P staging

    const int tid  = threadIdx.x;
    const int lane = tid & 31;
    const int wib  = tid >> 5;
    const int bh   = blockIdx.x >> 4;     // 16 blocks per (b,h)
    const int qgrp = blockIdx.x & 15;
    const int h = bh % Hh, b = bh / Hh;
    const int m0   = (qgrp * 8 + wib) * 16;
    const int half = lane >> 4;
    const int ncol = lane & 15;
    const int Dm   = Hh * 64;

    const float sigma  = __expf(log_sigma[0]);
    const float inv2s2 = 1.0f / (2.0f * sigma * sigma);

    const _Float16* Qb = Qr + (size_t)bh * Lx * 64;
    const _Float16* Kb = Kr + (size_t)bh * Lx * 64;
    const _Float16* Vb = Vt + (size_t)bh * 64 * Lx;

    auto fillKV = [&](int p, int j0) {
        int kr = tid >> 3, kc = (tid & 7) * 8;          // 32 rows x 8 b128 chunks
        async_load_b128(&smK[p][kr * 64 + kc],
                        Kb + (size_t)(j0 + kr) * 64 + kc);
        int vr = tid >> 2, vc = (tid & 3) * 8;          // 64 rows x 4 b128 chunks
        async_load_b128(&smV[p][vr * 32 + vc],
                        Vb + (size_t)vr * Lx + j0 + vc);
    };

    v16h aQ0 = load_a_frag(Qb + (size_t)m0 * 64 + 0,  64, lane);
    v16h aQ1 = load_a_frag(Qb + (size_t)m0 * 64 + 32, 64, lane);

    float rmax[8], rsum[8];
#pragma unroll
    for (int v = 0; v < 8; ++v) { rmax[v] = -1e30f; rsum[v] = 0.f; }
    v8f accO[4] = {{}, {}, {}, {}};

    _Float16* pbuf = &smP[wib][0];
    const float qi_base = (float)(m0 + 8 * half);

    fillKV(0, 0);
    int par = 0;
    for (int j0 = 0; j0 < Lx; j0 += 32) {
        const bool pre = (j0 + 32) < Lx;
        if (pre) { fillKV(par ^ 1, j0 + 32); wait_async2(); }
        else     { wait_async0(); }
        __syncthreads();

        v8f S0 = {}, S1 = {};
        {
            v16h b0 = load_b_frag(&smK[par][0],        64, lane);
            v16h b1 = load_b_frag(&smK[par][32],       64, lane);
            S0 = WMMA_F16(aQ0, b0, S0);
            S0 = WMMA_F16(aQ1, b1, S0);
            v16h b2 = load_b_frag(&smK[par][16 * 64],      64, lane);
            v16h b3 = load_b_frag(&smK[par][16 * 64 + 32], 64, lane);
            S1 = WMMA_F16(aQ0, b2, S1);
            S1 = WMMA_F16(aQ1, b3, S1);
        }
        // online softmax over the 32 new columns (rows live in 16-lane halves)
#pragma unroll
        for (int v = 0; v < 8; ++v) {
            float ipos = qi_base + (float)v;
            float d0 = (float)(j0 + ncol) - ipos;
            float d1 = d0 + 16.0f;
            float b0v = __logf(__expf(-d0 * d0 * inv2s2) + 1e-6f);
            float b1v = __logf(__expf(-d1 * d1 * inv2s2) + 1e-6f);
            float s0 = S0[v] * 0.125f + b0v;
            float s1 = S1[v] * 0.125f + b1v;
            float mx = fmaxf(s0, s1);
            mx = fmaxf(mx, __shfl_xor(mx, 1, 32));
            mx = fmaxf(mx, __shfl_xor(mx, 2, 32));
            mx = fmaxf(mx, __shfl_xor(mx, 4, 32));
            mx = fmaxf(mx, __shfl_xor(mx, 8, 32));
            float mn    = fmaxf(rmax[v], mx);
            float alpha = __expf(rmax[v] - mn);
            float p0 = __expf(s0 - mn);
            float p1 = __expf(s1 - mn);
            float ps = p0 + p1;
            ps += __shfl_xor(ps, 1, 32);
            ps += __shfl_xor(ps, 2, 32);
            ps += __shfl_xor(ps, 4, 32);
            ps += __shfl_xor(ps, 8, 32);
            rsum[v] = rsum[v] * alpha + ps;
            rmax[v] = mn;
            accO[0][v] *= alpha; accO[1][v] *= alpha;
            accO[2][v] *= alpha; accO[3][v] *= alpha;
            pbuf[(v + 8 * half) * 32 + ncol]      = (_Float16)p0;
            pbuf[(v + 8 * half) * 32 + 16 + ncol] = (_Float16)p1;
        }
        // same-wave LDS ops are in-order; drain DScnt before re-reading P
        asm volatile("s_wait_dscnt 0x0" ::: "memory");
        v16h aP = load_a_frag(pbuf, 32, lane);
#pragma unroll
        for (int c = 0; c < 4; ++c) {
            v16h bv = load_b_frag(&smV[par][(c * 16) * 32], 32, lane);
            accO[c] = WMMA_F16(aP, bv, accO[c]);
        }
        __syncthreads();                  // tiles consumed before next overwrite
        par ^= 1;
    }
#pragma unroll
    for (int c = 0; c < 4; ++c)
#pragma unroll
        for (int v = 0; v < 8; ++v) {
            int lq = m0 + v + 8 * half;
            int dv = c * 16 + ncol;
            float o = accO[c][v] / rsum[v];
            O[((size_t)b * Lx + lq) * Dm + h * 64 + dv] = (_Float16)o;
        }
}

// ------------------------------------------------------------------- launcher
extern "C" void kernel_launch(void* const* d_in, const int* in_sizes, int n_in,
                              void* d_out, int out_size, void* d_ws, size_t ws_size,
                              hipStream_t stream) {
    (void)in_sizes; (void)n_in; (void)out_size; (void)ws_size;
    const int B = 2, L = 2048, D = 1024, H = 16;
    const int M = B * L;

    const float* x       = (const float*)d_in[0];
    const int*   pos     = (const int*)d_in[1];
    const float* Wq      = (const float*)d_in[2];
    const float* bq      = (const float*)d_in[3];
    const float* Wk      = (const float*)d_in[4];
    const float* bk      = (const float*)d_in[5];
    const float* Wv      = (const float*)d_in[6];
    const float* bv      = (const float*)d_in[7];
    const float* Wo      = (const float*)d_in[8];
    const float* bo      = (const float*)d_in[9];
    const float* log_s   = (const float*)d_in[10];
    const float* Xg      = (const float*)d_in[11];
    const float* log_sig = (const float*)d_in[12];

    char* ws = (char*)d_ws;
    size_t off = 0;
    auto alloc = [&](size_t bytes) {
        char* p = ws + off;
        off += (bytes + 255) & ~(size_t)255;
        return p;
    };

    _Float16* Xh  = (_Float16*)alloc((size_t)M * D * 2);
    _Float16* Wqt = (_Float16*)alloc((size_t)D * D * 2);
    _Float16* Wkt = (_Float16*)alloc((size_t)D * D * 2);
    _Float16* Wvt = (_Float16*)alloc((size_t)D * D * 2);
    _Float16* Wot = (_Float16*)alloc((size_t)D * D * 2);
    float*    Qf  = (float*)alloc((size_t)M * D * 4);
    float*    Kf  = (float*)alloc((size_t)M * D * 4);
    float*    Vf  = (float*)alloc((size_t)M * D * 4);
    _Float16* Qrh = (_Float16*)alloc((size_t)M * D * 2);
    _Float16* Krh = (_Float16*)alloc((size_t)M * D * 2);
    _Float16* Vth = (_Float16*)alloc((size_t)M * D * 2);
    _Float16* Oh  = Xh;  // Xh dead after projection GEMMs; reuse

    const int n = M * D;
    cvt_f32_to_f16<<<(n + 255) / 256, 256, 0, stream>>>(x, Xh, n);

    dim3 tb(32, 8), tg(D / 32, D / 32);
    transpose_w_f16<<<tg, tb, 0, stream>>>(Wq, Wqt, D);
    transpose_w_f16<<<tg, tb, 0, stream>>>(Wk, Wkt, D);
    transpose_w_f16<<<tg, tb, 0, stream>>>(Wv, Wvt, D);
    transpose_w_f16<<<tg, tb, 0, stream>>>(Wo, Wot, D);

    // blocks = (N/64) * (M/128) = 16 * 32 = 512
    const int gemm_blocks = (D / 64) * (M / 128);
    gemm_f16_wmma<<<gemm_blocks, 256, 0, stream>>>(Xh, Wqt, bq, Qf, M, D, D);
    gemm_f16_wmma<<<gemm_blocks, 256, 0, stream>>>(Xh, Wkt, bk, Kf, M, D, D);
    gemm_f16_wmma<<<gemm_blocks, 256, 0, stream>>>(Xh, Wvt, bv, Vf, M, D, D);

    vt_transpose<<<(n + 255) / 256, 256, 0, stream>>>(Vf, Vth, B, L, H);

    cayley_apply<<<H * L, 64, 0, stream>>>(Xg, pos, log_s, Qf, Kf, Qrh, Krh, B, L, H);

    // blocks = B*H * (L/128) = 32 * 16 = 512
    const int attn_blocks = B * H * (L / 128);
    attn_wmma<<<attn_blocks, 256, 0, stream>>>(Qrh, Krh, Vth, log_sig, Oh, B, L, H);

    gemm_f16_wmma<<<gemm_blocks, 256, 0, stream>>>(Oh, Wot, bo, (float*)d_out, M, D, D);
}
